// Decoder_54537494725147
// MI455X (gfx1250) — compile-verified, matrix operand order checked by reading, not run on hardware
//
#include <hip/hip_runtime.h>
#include <hip/hip_bf16.h>
#include <stdint.h>

typedef __bf16 bf16_t;
typedef __attribute__((ext_vector_type(16))) __bf16 v16bf;
typedef __attribute__((ext_vector_type(8)))  float  v8f;

#define B_    64
#define PIX   196
#define ENC_  2048
#define DEC_  512
#define ATT_  512
#define EMB_  512
#define V_    10000
#define L_    52
#define T_    51
#define XDIM  2560   // EMB_ + ENC_

// ---------------------------------------------------------------------------
// WMMA helpers: bf16 16x16x32, f32 accumulate.
// Fragment layout per CDNA5 ISA (16-bit A 16x32): lane<16 holds K {0..7,16..23},
// lane>=16 holds K {8..15,24..31}; 2 bf16 per VGPR (packed 32-bit loads).
// Weights are stored output-major (Wt[n][k], k contiguous) so the B fragment
// uses the same contiguous packed load as A.
// ---------------------------------------------------------------------------
__device__ __forceinline__ v16bf load_frag(const bf16_t* __restrict__ row, int k0, int half) {
  union { v16bf v; uint32_t u[8]; } f;
#pragma unroll
  for (int i = 0; i < 8; ++i) {
    int k = k0 + ((i & 4) << 2) + half * 8 + (i & 3) * 2;   // (i<4?0:16)+half*8+(i%4)*2
    f.u[i] = *(const uint32_t*)(row + k);
  }
  return f.v;
}

__device__ __forceinline__ v8f wmma_bf16(v16bf a, v16bf b, v8f c) {
  return __builtin_amdgcn_wmma_f32_16x16x32_bf16(false, a, false, b, (short)0, c, false, false);
}

__device__ __forceinline__ float sigmoidf_(float x) { return 1.0f / (1.0f + __expf(-x)); }

// ---------------------------------------------------------------------------
// Setup kernels
// ---------------------------------------------------------------------------
__global__ void k_sort(const int* __restrict__ cap_len, const int* __restrict__ caps_in,
                       int* __restrict__ order_i, int* __restrict__ declen_i,
                       int* __restrict__ caps_s,
                       float* __restrict__ out_caps, float* __restrict__ out_declen,
                       float* __restrict__ out_order) {
  __shared__ int sh_len[B_];
  __shared__ int sh_order[B_];
  int i = threadIdx.x;
  if (i < B_) sh_len[i] = cap_len[i];
  __syncthreads();
  if (i < B_) {                       // stable argsort of -length (rank count)
    int li = sh_len[i], rank = 0;
    for (int j = 0; j < B_; ++j) {
      int lj = sh_len[j];
      if (lj > li || (lj == li && j < i)) rank++;
    }
    sh_order[rank] = i;
  }
  __syncthreads();
  if (i < B_) {
    int src = sh_order[i];
    order_i[i]    = src;
    int dl        = sh_len[src] - 1;
    declen_i[i]   = dl;
    out_order[i]  = (float)src;
    out_declen[i] = (float)dl;
    for (int t = 0; t < L_; ++t) {
      int cv = caps_in[src * L_ + t];
      caps_s[i * L_ + t]   = cv;
      out_caps[i * L_ + t] = (float)cv;
    }
  }
}

// transpose-convert fp32 (K,N) -> bf16 out[n*K + k]
__global__ void k_convert_T(const float* __restrict__ in, bf16_t* __restrict__ out,
                            int K, int N) {
  int idx = blockIdx.x * blockDim.x + threadIdx.x;
  if (idx >= K * N) return;
  int n = idx / K, k = idx % K;
  out[idx] = (bf16_t)in[(long)k * N + n];
}

// plain convert fp32 -> bf16 (already output-major, e.g. lstm_Wih/Whh)
__global__ void k_convert(const float* __restrict__ in, bf16_t* __restrict__ out, int count) {
  int idx = blockIdx.x * blockDim.x + threadIdx.x;
  if (idx < count) out[idx] = (bf16_t)in[idx];
}

__global__ void k_gather_enc(const float* __restrict__ enc, const int* __restrict__ order_i,
                             bf16_t* __restrict__ encs) {
  int idx = blockIdx.x * blockDim.x + threadIdx.x;      // < 64*196*2048
  int b = idx / (PIX * ENC_);
  int rest = idx - b * (PIX * ENC_);
  encs[idx] = (bf16_t)enc[(long)order_i[b] * (PIX * ENC_) + rest];
}

__global__ void k_mean(const float* __restrict__ enc, const int* __restrict__ order_i,
                       float* __restrict__ mean) {
  int idx = blockIdx.x * blockDim.x + threadIdx.x;      // b*2048 + e
  int b = idx >> 11, e = idx & (ENC_ - 1);
  const float* p = enc + (long)order_i[b] * (PIX * ENC_) + e;
  float s = 0.0f;
  for (int px = 0; px < PIX; ++px) s += p[px * ENC_];
  mean[idx] = s * (1.0f / PIX);
}

__global__ void k_emb(const float* __restrict__ embW, const int* __restrict__ caps_s,
                      bf16_t* __restrict__ embs) {
  int idx = blockIdx.x * blockDim.x + threadIdx.x;      // < 64*52*512
  int e = idx % EMB_;
  int bt = idx / EMB_;
  embs[idx] = (bf16_t)embW[(long)caps_s[bt] * EMB_ + e];
}

__global__ void k_init_hc(const float* __restrict__ mean,
                          const float* __restrict__ HW, const float* __restrict__ Hb,
                          const float* __restrict__ CW, const float* __restrict__ Cb,
                          float* __restrict__ h, float* __restrict__ c,
                          bf16_t* __restrict__ h_bf) {
  int idx = blockIdx.x * blockDim.x + threadIdx.x;      // 64*512
  int b = idx / DEC_, j = idx % DEC_;
  const float* m = mean + b * ENC_;
  float sh = Hb[j], sc = Cb[j];
  for (int e = 0; e < ENC_; ++e) {
    float mv = m[e];
    sh += mv * HW[e * DEC_ + j];
    sc += mv * CW[e * DEC_ + j];
  }
  h[idx] = sh; c[idx] = sc; h_bf[idx] = (bf16_t)sh;
}

// ---------------------------------------------------------------------------
// Generic bf16 WMMA GEMM: C[M,N] = A[M,K] * Wt[N,K]^T + bias, one 16x16 tile/wave
// ---------------------------------------------------------------------------
__global__ void __launch_bounds__(256)
k_wmma_gemm(const bf16_t* __restrict__ A, int lda,
            const bf16_t* __restrict__ W, int ldw,
            const float* __restrict__ bias,
            float* __restrict__ C, int ldc,
            int Ntiles, int K, int totTiles) {
  int tile = blockIdx.x * 8 + (threadIdx.x >> 5);
  if (tile >= totTiles) return;
  int lane = threadIdx.x & 31, half = lane >> 4, l15 = lane & 15;
  int mt = tile / Ntiles, nt = tile % Ntiles;
  const bf16_t* arow = A + (long)(mt * 16 + l15) * lda;
  const bf16_t* wrow = W + (long)(nt * 16 + l15) * ldw;
  v8f acc = {};
  for (int k0 = 0; k0 < K; k0 += 32)
    acc = wmma_bf16(load_frag(arow, k0, half), load_frag(wrow, k0, half), acc);
  int n = nt * 16 + l15;
  float bn = bias ? bias[n] : 0.0f;
#pragma unroll
  for (int i = 0; i < 8; ++i) {
    int m = mt * 16 + i + 8 * half;
    C[(long)m * ldc + n] = acc[i] + bn;
  }
}

// gate = sigmoid(h @ fbeta^T + b); gawe = bf16(gate * awe)  (M=64, N=2048, K=512)
__global__ void __launch_bounds__(256)
k_gate(const bf16_t* __restrict__ hb, const bf16_t* __restrict__ Wt,
       const float* __restrict__ bias, const float* __restrict__ awe,
       bf16_t* __restrict__ gawe) {
  int tile = blockIdx.x * 8 + (threadIdx.x >> 5);       // 512 tiles, 64 blocks
  int lane = threadIdx.x & 31, half = lane >> 4, l15 = lane & 15;
  int mt = tile >> 7, nt = tile & 127;
  const bf16_t* arow = hb + (mt * 16 + l15) * DEC_;
  const bf16_t* wrow = Wt + (nt * 16 + l15) * DEC_;
  v8f acc = {};
  for (int k0 = 0; k0 < DEC_; k0 += 32)
    acc = wmma_bf16(load_frag(arow, k0, half), load_frag(wrow, k0, half), acc);
  int n = nt * 16 + l15;
  float bn = bias[n];
#pragma unroll
  for (int i = 0; i < 8; ++i) {
    int m = mt * 16 + i + 8 * half;
    float g = sigmoidf_(acc[i] + bn);
    gawe[m * ENC_ + n] = (bf16_t)(g * awe[m * ENC_ + n]);
  }
}

// gates = [emb_t | gawe] @ Wih^T + h @ Whh^T + bih + bhh  (3 fused K-phases)
__global__ void __launch_bounds__(256)
k_lstm_gates(const bf16_t* __restrict__ embs, int t,
             const bf16_t* __restrict__ gawe, const bf16_t* __restrict__ hb,
             const bf16_t* __restrict__ Wih, const bf16_t* __restrict__ Whh,
             const float* __restrict__ bih, const float* __restrict__ bhh,
             float* __restrict__ gates) {
  int tile = blockIdx.x * 8 + (threadIdx.x >> 5);       // 512 tiles, 64 blocks
  int lane = threadIdx.x & 31, half = lane >> 4, l15 = lane & 15;
  int mt = tile >> 7, nt = tile & 127;
  int m15 = mt * 16 + l15, n15 = nt * 16 + l15;
  v8f acc = {};
  {
    const bf16_t* arow = embs + (long)m15 * (L_ * EMB_) + t * EMB_;
    const bf16_t* wrow = Wih + (long)n15 * XDIM;
    for (int k0 = 0; k0 < EMB_; k0 += 32)
      acc = wmma_bf16(load_frag(arow, k0, half), load_frag(wrow, k0, half), acc);
  }
  {
    const bf16_t* arow = gawe + m15 * ENC_;
    const bf16_t* wrow = Wih + (long)n15 * XDIM + EMB_;
    for (int k0 = 0; k0 < ENC_; k0 += 32)
      acc = wmma_bf16(load_frag(arow, k0, half), load_frag(wrow, k0, half), acc);
  }
  {
    const bf16_t* arow = hb + m15 * DEC_;
    const bf16_t* wrow = Whh + n15 * DEC_;
    for (int k0 = 0; k0 < DEC_; k0 += 32)
      acc = wmma_bf16(load_frag(arow, k0, half), load_frag(wrow, k0, half), acc);
  }
  float bn = bih[n15] + bhh[n15];
#pragma unroll
  for (int i = 0; i < 8; ++i) {
    int m = mt * 16 + i + 8 * half;
    gates[m * ENC_ + n15] = acc[i] + bn;
  }
}

__global__ void k_lstm_elem(const float* __restrict__ gates, const int* __restrict__ declen,
                            int t, float* __restrict__ h, float* __restrict__ c,
                            bf16_t* __restrict__ h_bf, bf16_t* __restrict__ hnew_bf) {
  int idx = blockIdx.x * blockDim.x + threadIdx.x;      // 64*512
  int b = idx / DEC_, j = idx % DEC_;
  const float* g = gates + b * 2048;
  float ig = sigmoidf_(g[j]);
  float fg = sigmoidf_(g[j + 512]);
  float gg = tanhf(g[j + 1024]);
  float og = sigmoidf_(g[j + 1536]);
  float c_new = fg * c[idx] + ig * gg;
  float h_new = og * tanhf(c_new);
  hnew_bf[idx] = (bf16_t)h_new;                         // pre-mask h_new feeds fc
  if (t < declen[b]) { h[idx] = h_new; c[idx] = c_new; }
  h_bf[idx] = (bf16_t)h[idx];                           // masked h feeds next step
}

// preds[b][t][:] = mask ? hnew @ fc_W + fc_b : 0   (M=64, N=10000, K=512)
__global__ void __launch_bounds__(256)
k_fc(const bf16_t* __restrict__ hnb, const bf16_t* __restrict__ Wt,
     const float* __restrict__ bias, const int* __restrict__ declen,
     int t, float* __restrict__ preds) {
  int tile = blockIdx.x * 8 + (threadIdx.x >> 5);       // 4*625 = 2500 tiles
  if (tile >= 2500) return;
  int lane = threadIdx.x & 31, half = lane >> 4, l15 = lane & 15;
  int mt = tile / 625, nt = tile % 625;
  const bf16_t* arow = hnb + (mt * 16 + l15) * DEC_;
  const bf16_t* wrow = Wt + (long)(nt * 16 + l15) * DEC_;
  v8f acc = {};
  for (int k0 = 0; k0 < DEC_; k0 += 32)
    acc = wmma_bf16(load_frag(arow, k0, half), load_frag(wrow, k0, half), acc);
  int n = nt * 16 + l15;
  float bn = bias[n];
#pragma unroll
  for (int i = 0; i < 8; ++i) {
    int m = mt * 16 + i + 8 * half;
    float v = (t < declen[m]) ? (acc[i] + bn) : 0.0f;
    preds[((long)m * T_ + t) * V_ + n] = v;
  }
}

// attention: e = relu(att1 + att2) @ w + b; alpha = softmax(e); awe = alpha @ enc
__global__ void __launch_bounds__(256)
k_attention(const float* __restrict__ att1, const float* __restrict__ att2,
            const float* __restrict__ full_w, const float* __restrict__ full_b,
            const bf16_t* __restrict__ encs, const int* __restrict__ declen,
            int t, float* __restrict__ awe, float* __restrict__ alphas_out) {
  int b = blockIdx.x;
  int tid = threadIdx.x;
  __shared__ float sh_a[PIX];
  __shared__ float red[256];
  const float* a2 = att2 + b * ATT_;
  for (int p = tid; p < PIX; p += 256) {
    const float* a1 = att1 + (long)(b * PIX + p) * ATT_;
    float e = full_b[0];
    for (int k = 0; k < ATT_; ++k) {
      float r = a1[k] + a2[k];
      e += (r > 0.0f ? r : 0.0f) * full_w[k];
    }
    sh_a[p] = e;
  }
  __syncthreads();
  float lm = -1e30f;
  for (int p = tid; p < PIX; p += 256) lm = fmaxf(lm, sh_a[p]);
  red[tid] = lm; __syncthreads();
  for (int s = 128; s > 0; s >>= 1) { if (tid < s) red[tid] = fmaxf(red[tid], red[tid + s]); __syncthreads(); }
  float mx = red[0]; __syncthreads();
  float ls = 0.0f;
  for (int p = tid; p < PIX; p += 256) ls += __expf(sh_a[p] - mx);
  red[tid] = ls; __syncthreads();
  for (int s = 128; s > 0; s >>= 1) { if (tid < s) red[tid] += red[tid + s]; __syncthreads(); }
  float inv = 1.0f / red[0];
  bool mask = t < declen[b];
  __syncthreads();
  for (int p = tid; p < PIX; p += 256) {
    float al = __expf(sh_a[p] - mx) * inv;
    sh_a[p] = al;
    alphas_out[((long)b * T_ + t) * PIX + p] = mask ? al : 0.0f;
  }
  __syncthreads();
  const bf16_t* eb = encs + (long)b * PIX * ENC_;
  for (int e0 = tid; e0 < ENC_; e0 += 256) {
    float s = 0.0f;
    for (int p = 0; p < PIX; ++p) s += sh_a[p] * (float)eb[p * ENC_ + e0];
    awe[b * ENC_ + e0] = s;
  }
}

// ---------------------------------------------------------------------------
extern "C" void kernel_launch(void* const* d_in, const int* in_sizes, int n_in,
                              void* d_out, int out_size, void* d_ws, size_t ws_size,
                              hipStream_t stream) {
  const float* enc0    = (const float*)d_in[0];
  const int*   caps0   = (const int*)d_in[1];
  const int*   lens0   = (const int*)d_in[2];
  const float* embW    = (const float*)d_in[3];
  const float* attEncW = (const float*)d_in[4];
  const float* attEncB = (const float*)d_in[5];
  const float* attDecW = (const float*)d_in[6];
  const float* attDecB = (const float*)d_in[7];
  const float* fullW   = (const float*)d_in[8];
  const float* fullB   = (const float*)d_in[9];
  const float* WihF    = (const float*)d_in[10];
  const float* WhhF    = (const float*)d_in[11];
  const float* bih     = (const float*)d_in[12];
  const float* bhh     = (const float*)d_in[13];
  const float* HW      = (const float*)d_in[14];
  const float* Hb      = (const float*)d_in[15];
  const float* CW      = (const float*)d_in[16];
  const float* Cb      = (const float*)d_in[17];
  const float* fbW     = (const float*)d_in[18];
  const float* fbB     = (const float*)d_in[19];
  const float* fcW     = (const float*)d_in[20];
  const float* fcB     = (const float*)d_in[21];

  float* out        = (float*)d_out;
  float* out_preds  = out;
  float* out_caps   = out + (size_t)B_ * T_ * V_;
  float* out_declen = out_caps + B_ * L_;
  float* out_alphas = out_declen + B_;
  float* out_order  = out_alphas + (size_t)B_ * T_ * PIX;

  char* ws = (char*)d_ws;
  size_t off = 0;
  auto alloc = [&](size_t bytes) -> void* {
    void* p = ws + off;
    off += (bytes + 255) & ~(size_t)255;
    return p;
  };
  int*    order_i  = (int*)alloc(B_ * sizeof(int));
  int*    declen_i = (int*)alloc(B_ * sizeof(int));
  int*    caps_s   = (int*)alloc((size_t)B_ * L_ * sizeof(int));
  bf16_t* encs     = (bf16_t*)alloc((size_t)B_ * PIX * ENC_ * 2);
  float*  att1     = (float*)alloc((size_t)B_ * PIX * ATT_ * 4);
  bf16_t* embs     = (bf16_t*)alloc((size_t)B_ * L_ * EMB_ * 2);
  float*  meanb    = (float*)alloc((size_t)B_ * ENC_ * 4);
  bf16_t* wAttEncT = (bf16_t*)alloc((size_t)ATT_ * ENC_ * 2);   // [n<512][k<2048]
  bf16_t* wAttDecT = (bf16_t*)alloc((size_t)ATT_ * DEC_ * 2);   // [n<512][k<512]
  bf16_t* wFbetaT  = (bf16_t*)alloc((size_t)ENC_ * DEC_ * 2);   // [n<2048][k<512]
  bf16_t* wFcT     = (bf16_t*)alloc((size_t)V_ * DEC_ * 2);     // [n<10000][k<512]
  bf16_t* wWih     = (bf16_t*)alloc((size_t)ENC_ * XDIM * 2);   // already output-major
  bf16_t* wWhh     = (bf16_t*)alloc((size_t)ENC_ * DEC_ * 2);
  float*  h        = (float*)alloc((size_t)B_ * DEC_ * 4);
  float*  c        = (float*)alloc((size_t)B_ * DEC_ * 4);
  bf16_t* h_bf     = (bf16_t*)alloc((size_t)B_ * DEC_ * 2);
  bf16_t* hnew_bf  = (bf16_t*)alloc((size_t)B_ * DEC_ * 2);
  float*  att2     = (float*)alloc((size_t)B_ * ATT_ * 4);
  float*  awe      = (float*)alloc((size_t)B_ * ENC_ * 4);
  bf16_t* gawe     = (bf16_t*)alloc((size_t)B_ * ENC_ * 2);
  float*  gates    = (float*)alloc((size_t)B_ * ENC_ * 4);

  // ---- setup (once per call) ----
  k_sort<<<1, 64, 0, stream>>>(lens0, caps0, order_i, declen_i, caps_s,
                               out_caps, out_declen, out_order);
  k_convert_T<<<(ENC_ * ATT_ + 255) / 256, 256, 0, stream>>>(attEncW, wAttEncT, ENC_, ATT_);
  k_convert_T<<<(DEC_ * ATT_ + 255) / 256, 256, 0, stream>>>(attDecW, wAttDecT, DEC_, ATT_);
  k_convert_T<<<(DEC_ * ENC_ + 255) / 256, 256, 0, stream>>>(fbW, wFbetaT, DEC_, ENC_);
  k_convert_T<<<(DEC_ * V_ + 255) / 256, 256, 0, stream>>>(fcW, wFcT, DEC_, V_);
  k_convert<<<(ENC_ * XDIM + 255) / 256, 256, 0, stream>>>(WihF, wWih, ENC_ * XDIM);
  k_convert<<<(ENC_ * DEC_ + 255) / 256, 256, 0, stream>>>(WhhF, wWhh, ENC_ * DEC_);
  k_gather_enc<<<(B_ * PIX * ENC_) / 256, 256, 0, stream>>>(enc0, order_i, encs);
  k_mean<<<(B_ * ENC_) / 256, 256, 0, stream>>>(enc0, order_i, meanb);
  k_emb<<<(B_ * L_ * EMB_) / 256, 256, 0, stream>>>(embW, caps_s, embs);
  k_init_hc<<<(B_ * DEC_) / 256, 256, 0, stream>>>(meanb, HW, Hb, CW, Cb, h, c, h_bf);
  // att1 = enc @ att_enc_W + b : (12544 x 2048) x (2048 x 512), 784*32 tiles
  k_wmma_gemm<<<(784 * 32) / 8, 256, 0, stream>>>(encs, ENC_, wAttEncT, ENC_, attEncB,
                                                  att1, ATT_, 32, ENC_, 784 * 32);

  // ---- 51 sequential LSTM steps ----
  for (int t = 0; t < T_; ++t) {
    // att2 = h @ att_dec_W + b : 4*32 = 128 tiles
    k_wmma_gemm<<<128 / 8, 256, 0, stream>>>(h_bf, DEC_, wAttDecT, DEC_, attDecB,
                                             att2, ATT_, 32, DEC_, 128);
    k_attention<<<B_, 256, 0, stream>>>(att1, att2, fullW, fullB, encs, declen_i,
                                        t, awe, out_alphas);
    k_gate<<<(4 * 128) / 8, 256, 0, stream>>>(h_bf, wFbetaT, fbB, awe, gawe);
    k_lstm_gates<<<(4 * 128) / 8, 256, 0, stream>>>(embs, t, gawe, h_bf, wWih, wWhh,
                                                    bih, bhh, gates);
    k_lstm_elem<<<(B_ * DEC_) / 256, 256, 0, stream>>>(gates, declen_i, t, h, c,
                                                       h_bf, hnew_bf);
    k_fc<<<(2500 + 7) / 8, 256, 0, stream>>>(hnew_bf, wFcT, fcB, declen_i, t, out_preds);
  }
}
